// InteractionBlock_27908697489889
// MI455X (gfx1250) — compile-verified, matrix operand order checked by reading, not run on hardware
//
#include <hip/hip_runtime.h>
#include <hip/hip_bf16.h>

typedef _Float16 v8h  __attribute__((ext_vector_type(8)));
typedef _Float16 v16h __attribute__((ext_vector_type(16)));
typedef float    v8f  __attribute__((ext_vector_type(8)));

#define KDIM 768
#define LDA  776          // padded LDS row stride (halfs) to spread banks
#define HEADS 6

// ---------------------------------------------------------------------------
// Convert W[K=768, N] (f32, row-major K) -> Wt[N16][768] f16, zero-padded rows
// ---------------------------------------------------------------------------
__global__ __launch_bounds__(256) void convert_wt(const float* __restrict__ W,
                                                  _Float16* __restrict__ Wt,
                                                  int N, int N16) {
    int idx = blockIdx.x * 256 + threadIdx.x;
    int total = N16 * KDIM;
    if (idx >= total) return;
    int n = idx / KDIM, k = idx - n * KDIM;
    float v = (n < N) ? W[(size_t)k * N + n] : 0.f;
    Wt[idx] = (_Float16)v;
}

// ---------------------------------------------------------------------------
// out[M,N] = A[M,768] @ W[768,N] + bias, via v_wmma_f32_16x16x32_f16.
// Wt is f16 [N16,768] (transposed, N16 multiple of 64, zero-padded).
// Block = 256 thr (8 waves). Each wave owns a 16x64 C strip (4 accumulators),
// reusing one A fragment across 4 WMMAs per k-step. A tile (16x768) staged
// once per block in LDS as f16.
// ---------------------------------------------------------------------------
__global__ __launch_bounds__(256) void gemm_wmma(const float* __restrict__ A,
                                                 const _Float16* __restrict__ Wt,
                                                 const float* __restrict__ bias,
                                                 float* __restrict__ out,
                                                 int M, int N, int N16) {
    __shared__ _Float16 aT[16 * LDA];
    const int tid = threadIdx.x;
    const int m0  = blockIdx.x * 16;

    // cooperative load A tile 16x768 f32 -> f16 LDS (48 elems / thread)
    #pragma unroll 4
    for (int i = 0; i < 48; ++i) {
        int idx = i * 256 + tid;
        int r = idx / KDIM, k = idx - r * KDIM;
        int gr = m0 + r;
        float v = (gr < M) ? A[(size_t)gr * KDIM + k] : 0.f;
        aT[r * LDA + k] = (_Float16)v;
    }
    __syncthreads();

    const int wave = tid >> 5, lane = tid & 31;
    const int n0 = blockIdx.y * 512 + wave * 64;
    if (n0 >= N16) return;                         // wave-uniform exit

    const int  nl   = lane & 15, half = lane >> 4;
    const _Float16* arow = &aT[nl * LDA];          // A: m = lane&15
    const _Float16* bp[4];
    #pragma unroll
    for (int t = 0; t < 4; ++t)                    // B: n = n0 + t*16 + lane&15
        bp[t] = Wt + (size_t)(n0 + t * 16 + nl) * KDIM;

    v8f c0 = {}, c1 = {}, c2 = {}, c3 = {};
    #pragma unroll 2
    for (int kk = 0; kk < KDIM; kk += 32) {
        const int base = kk + 8 * half;
        v8h a0 = *(const v8h*)(arow + base);        // ds_load_b128 x2
        v8h a1 = *(const v8h*)(arow + base + 16);
        v16h a;
        #pragma unroll
        for (int i = 0; i < 8; ++i) { a[i] = a0[i]; a[i + 8] = a1[i]; }

        v16h b;
        v8h b0, b1;
        #define BFRAG(T)                                                     \
            b0 = *(const v8h*)(bp[T] + base);                                \
            b1 = *(const v8h*)(bp[T] + base + 16);                           \
            _Pragma("unroll")                                                \
            for (int i = 0; i < 8; ++i) { b[i] = b0[i]; b[i + 8] = b1[i]; }
        BFRAG(0)
        c0 = __builtin_amdgcn_wmma_f32_16x16x32_f16(false, a, false, b, (short)0, c0, false, false);
        BFRAG(1)
        c1 = __builtin_amdgcn_wmma_f32_16x16x32_f16(false, a, false, b, (short)0, c1, false, false);
        BFRAG(2)
        c2 = __builtin_amdgcn_wmma_f32_16x16x32_f16(false, a, false, b, (short)0, c2, false, false);
        BFRAG(3)
        c3 = __builtin_amdgcn_wmma_f32_16x16x32_f16(false, a, false, b, (short)0, c3, false, false);
        #undef BFRAG
    }

    // C/D layout: lane%16 = n, VGPR v -> m = v + 8*(lane>>4)
    v8f cs[4] = {c0, c1, c2, c3};
    #pragma unroll
    for (int t = 0; t < 4; ++t) {
        int n = n0 + t * 16 + nl;
        if (n < N) {
            float bv = bias ? bias[n] : 0.f;
            #pragma unroll
            for (int v = 0; v < 8; ++v) {
                int m = m0 + v + 8 * half;
                if (m < M) out[(size_t)m * N + n] = cs[t][v] + bv;
            }
        }
    }
}

// ---------------------------------------------------------------------------
// LayerNorm over 768, optional residual add:  out = LN(X (+ Y)) * g + b
// one row per 256-thread block
// ---------------------------------------------------------------------------
__global__ __launch_bounds__(256) void layernorm_add(const float* __restrict__ X,
                                                     const float* __restrict__ Y,
                                                     const float* __restrict__ g,
                                                     const float* __restrict__ b,
                                                     float* __restrict__ out, int M) {
    __shared__ float sbuf[256];
    int row = blockIdx.x;
    if (row >= M) return;
    const float* x = X + (size_t)row * KDIM;
    const float* y = Y ? Y + (size_t)row * KDIM : nullptr;
    float v[3], s = 0.f;
    #pragma unroll
    for (int i = 0; i < 3; ++i) {
        int c = i * 256 + threadIdx.x;
        v[i] = x[c] + (y ? y[c] : 0.f);
        s += v[i];
    }
    sbuf[threadIdx.x] = s; __syncthreads();
    for (int off = 128; off > 0; off >>= 1) {
        if (threadIdx.x < off) sbuf[threadIdx.x] += sbuf[threadIdx.x + off];
        __syncthreads();
    }
    float mean = sbuf[0] * (1.f / KDIM);
    __syncthreads();
    float s2 = 0.f;
    #pragma unroll
    for (int i = 0; i < 3; ++i) { float d = v[i] - mean; s2 += d * d; }
    sbuf[threadIdx.x] = s2; __syncthreads();
    for (int off = 128; off > 0; off >>= 1) {
        if (threadIdx.x < off) sbuf[threadIdx.x] += sbuf[threadIdx.x + off];
        __syncthreads();
    }
    float r = rsqrtf(sbuf[0] * (1.f / KDIM) + 1e-6f);
    #pragma unroll
    for (int i = 0; i < 3; ++i) {
        int c = i * 256 + threadIdx.x;
        out[(size_t)row * KDIM + c] = (v[i] - mean) * r * g[c] + b[c];
    }
}

// ---------------------------------------------------------------------------
// softmax over trailing dim lp (<=12), one row per thread
// ---------------------------------------------------------------------------
__global__ __launch_bounds__(256) void softmax_rows(float* __restrict__ x,
                                                    int rows, int lp) {
    int r = blockIdx.x * 256 + threadIdx.x;
    if (r >= rows) return;
    float* p = x + (size_t)r * lp;
    float e[12], mx = -1e30f, s = 0.f;
    for (int i = 0; i < lp; ++i) mx = fmaxf(mx, p[i]);
    for (int i = 0; i < lp; ++i) { e[i] = __expf(p[i] - mx); s += e[i]; }
    float inv = 1.f / s;
    for (int i = 0; i < lp; ++i) p[i] = e[i] * inv;
}

// ---------------------------------------------------------------------------
// Multi-scale deformable sampling. One wave per (q, head); lane owns 4 chans.
// vproj: [Ntok, 768] (chan = h*128+d)   off: [Lq, 6*L*4*2]   aw: [Lq, 6*L*4]
// ref:   [Lq, 2]                         out: [Lq, 768]
// ---------------------------------------------------------------------------
__global__ __launch_bounds__(256) void deform_sample(const float* __restrict__ vproj,
                                                     const float* __restrict__ off,
                                                     const float* __restrict__ aw,
                                                     const float* __restrict__ ref,
                                                     float* __restrict__ out,
                                                     int Lq, int L,
                                                     int h0, int w0, int s0,
                                                     int h1, int w1, int s1,
                                                     int h2, int w2, int s2) {
    int wid  = (blockIdx.x * 256 + threadIdx.x) >> 5;
    int lane = threadIdx.x & 31;
    int q = wid / HEADS, h = wid - q * HEADS;
    if (q >= Lq) return;

    int Hl_[3] = {h0, h1, h2}, Wl_[3] = {w0, w1, w2}, St_[3] = {s0, s1, s2};
    float rx = ref[(size_t)q * 2 + 0], ry = ref[(size_t)q * 2 + 1];
    int d0 = h * 128 + lane * 4;
    float4 acc = {0.f, 0.f, 0.f, 0.f};

    for (int l = 0; l < L; ++l) {
        int Hl = Hl_[l], Wl = Wl_[l], st = St_[l];
        for (int p = 0; p < 4; ++p) {
            int oi = (h * L + l) * 4 + p;
            float ox  = off[(size_t)q * (HEADS * L * 8) + oi * 2 + 0];
            float oy  = off[(size_t)q * (HEADS * L * 8) + oi * 2 + 1];
            float wgt = aw [(size_t)q * (HEADS * L * 4) + oi];
            // loc = ref + off/[W,H]; ix = loc.x*W - 0.5
            float ix = rx * (float)Wl + ox - 0.5f;
            float iy = ry * (float)Hl + oy - 0.5f;
            float x0f = floorf(ix), y0f = floorf(iy);
            float wx = ix - x0f, wy = iy - y0f;
            int x0 = (int)x0f, y0 = (int)y0f;

            auto gat = [&](int xi, int yi) -> float4 {
                float4 r = {0.f, 0.f, 0.f, 0.f};
                if (xi >= 0 && xi < Wl && yi >= 0 && yi < Hl)
                    r = *(const float4*)(vproj + (size_t)(st + yi * Wl + xi) * KDIM + d0);
                return r;
            };
            float4 v00 = gat(x0, y0),     v01 = gat(x0 + 1, y0);
            float4 v10 = gat(x0, y0 + 1), v11 = gat(x0 + 1, y0 + 1);
            float c00 = (1.f - wx) * (1.f - wy) * wgt, c01 = wx * (1.f - wy) * wgt;
            float c10 = (1.f - wx) * wy * wgt,         c11 = wx * wy * wgt;
            acc.x += c00 * v00.x + c01 * v01.x + c10 * v10.x + c11 * v11.x;
            acc.y += c00 * v00.y + c01 * v01.y + c10 * v10.y + c11 * v11.y;
            acc.z += c00 * v00.z + c01 * v01.z + c10 * v10.z + c11 * v11.z;
            acc.w += c00 * v00.w + c01 * v01.w + c10 * v10.w + c11 * v11.w;
        }
    }
    *(float4*)(out + (size_t)q * KDIM + d0) = acc;
}

// ---------------------------------------------------------------------------
// vit_out = q + gamma[c] * attn
// ---------------------------------------------------------------------------
__global__ __launch_bounds__(256) void residual_gamma(const float* __restrict__ q,
                                                      const float* __restrict__ attn,
                                                      const float* __restrict__ gamma,
                                                      float* __restrict__ out, int n) {
    int i = blockIdx.x * 256 + threadIdx.x;
    if (i >= n) return;
    int c = i % KDIM;
    out[i] = q[i] + gamma[c] * attn[i];
}

// ===========================================================================
extern "C" void kernel_launch(void* const* d_in, const int* in_sizes, int n_in,
                              void* d_out, int out_size, void* d_ws, size_t ws_size,
                              hipStream_t stream) {
    (void)in_sizes; (void)n_in; (void)out_size; (void)ws_size;

    const int LQ = 2304, NA = 12096, C = 768;

    const float* vit   = (const float*)d_in[0];   // [2304,768]
    const float* adp   = (const float*)d_in[1];   // [12096,768]
    const float* ref1  = (const float*)d_in[2];   // [2304,2]
    const float* ref2  = (const float*)d_in[3];   // [12096,2]
    const float* iqn_g = (const float*)d_in[4];
    const float* iqn_b = (const float*)d_in[5];
    const float* iWv   = (const float*)d_in[6];
    const float* ibv   = (const float*)d_in[7];
    const float* iWs   = (const float*)d_in[8];   // [768,144]
    const float* ibs   = (const float*)d_in[9];
    const float* iWa   = (const float*)d_in[10];  // [768,72]
    const float* iba   = (const float*)d_in[11];
    const float* iWo   = (const float*)d_in[12];
    const float* ibo   = (const float*)d_in[13];
    const float* igam  = (const float*)d_in[14];
    const float* evn_g = (const float*)d_in[15];
    const float* evn_b = (const float*)d_in[16];
    const float* eqn_g = (const float*)d_in[17];
    const float* eqn_b = (const float*)d_in[18];
    const float* eWv   = (const float*)d_in[19];
    const float* ebv   = (const float*)d_in[20];
    const float* eWs   = (const float*)d_in[21];  // [768,48]
    const float* ebs   = (const float*)d_in[22];
    const float* eWa   = (const float*)d_in[23];  // [768,24]
    const float* eba   = (const float*)d_in[24];
    const float* eWo   = (const float*)d_in[25];
    const float* ebo   = (const float*)d_in[26];
    // d_in[27]=H, d_in[28]=W (48, hardcoded)

    float* out_vit = (float*)d_out;               // [2304,768]
    float* out_adp = out_vit + (size_t)LQ * C;    // [12096,768]

    // -------- workspace layout (f32 then f16) --------
    float* f = (float*)d_ws;
    size_t o = 0;
    float* qn    = f + o; o += (size_t)LQ * C;          // reused as samp1
    float* vp1   = f + o; o += (size_t)NA * C;          // reused as samp2
    float* off1  = f + o; o += (size_t)LQ * 144;
    float* aw1   = f + o; o += (size_t)LQ * 72;
    float* attn1 = f + o; o += (size_t)LQ * C;
    float* val2  = f + o; o += (size_t)LQ * C;
    float* vp2   = f + o; o += (size_t)LQ * C;
    float* off2  = f + o; o += (size_t)NA * 48;
    float* aw2   = f + o; o += (size_t)NA * 24;
    float* attn2 = f + o; o += (size_t)NA * C;
    _Float16* hb = (_Float16*)(f + o);
    size_t ho = 0;
    _Float16* wv1t = hb + ho; ho += (size_t)768 * C;
    _Float16* ws1t = hb + ho; ho += (size_t)192 * C;   // 144 padded to 192
    _Float16* wa1t = hb + ho; ho += (size_t)128 * C;   // 72  padded to 128
    _Float16* wo1t = hb + ho; ho += (size_t)768 * C;
    _Float16* wv2t = hb + ho; ho += (size_t)768 * C;
    _Float16* ws2t = hb + ho; ho += (size_t)64  * C;   // 48  padded to 64
    _Float16* wa2t = hb + ho; ho += (size_t)64  * C;   // 24  padded to 64
    _Float16* wo2t = hb + ho; ho += (size_t)768 * C;

    auto cvt = [&](const float* W, _Float16* Wt, int N, int N16) {
        int total = N16 * C;
        convert_wt<<<(total + 255) / 256, 256, 0, stream>>>(W, Wt, N, N16);
    };
    auto gemm = [&](const float* A, const _Float16* Wt, const float* bias,
                    float* out, int M, int N, int N16) {
        dim3 grid((M + 15) / 16, (N16 + 511) / 512);
        gemm_wmma<<<grid, 256, 0, stream>>>(A, Wt, bias, out, M, N, N16);
    };

    // ---- weight conversion (f32 [K,N] -> f16 [N16,K], N16 mult of 64) ----
    cvt(iWv, wv1t, 768, 768);  cvt(iWs, ws1t, 144, 192);
    cvt(iWa, wa1t,  72, 128);  cvt(iWo, wo1t, 768, 768);
    cvt(eWv, wv2t, 768, 768);  cvt(eWs, ws2t,  48,  64);
    cvt(eWa, wa2t,  24,  64);  cvt(eWo, wo2t, 768, 768);

    // ======== Injector ========
    layernorm_add<<<LQ, 256, 0, stream>>>(vit, nullptr, iqn_g, iqn_b, qn, LQ);
    gemm(adp, wv1t, ibv, vp1, NA, 768, 768);                  // value proj
    gemm(qn,  ws1t, ibs, off1, LQ, 144, 192);                 // sampling offsets
    gemm(qn,  wa1t, iba, aw1,  LQ,  72, 128);                 // attn weights
    softmax_rows<<<(LQ * HEADS + 255) / 256, 256, 0, stream>>>(aw1, LQ * HEADS, 12);
    {   // sample into samp1 (aliases qn, dead now)
        int waves = LQ * HEADS;
        deform_sample<<<(waves * 32 + 255) / 256, 256, 0, stream>>>(
            vp1, off1, aw1, ref1, qn, LQ, 3,
            96, 96, 0, 48, 48, 9216, 24, 24, 11520);
    }
    gemm(qn, wo1t, ibo, attn1, LQ, 768, 768);                 // output proj
    residual_gamma<<<((LQ * C) + 255) / 256, 256, 0, stream>>>(
        vit, attn1, igam, out_vit, LQ * C);

    // ======== Extractor ========
    layernorm_add<<<LQ, 256, 0, stream>>>(out_vit, nullptr, evn_g, evn_b, val2, LQ);
    gemm(val2, wv2t, ebv, vp2, LQ, 768, 768);                 // value proj
    gemm(adp,  ws2t, ebs, off2, NA, 48, 64);                  // sampling offsets
    gemm(adp,  wa2t, eba, aw2,  NA, 24, 64);                  // attn weights
    softmax_rows<<<(NA * HEADS + 255) / 256, 256, 0, stream>>>(aw2, NA * HEADS, 4);
    {   // sample into samp2 (aliases vp1, dead now)
        int waves = NA * HEADS;
        deform_sample<<<(waves * 32 + 255) / 256, 256, 0, stream>>>(
            vp2, off2, aw2, ref2, vp1, NA, 1,
            48, 48, 0, 0, 0, 0, 0, 0, 0);
    }
    gemm(vp1, wo2t, ebo, attn2, NA, 768, 768);                // output proj
    layernorm_add<<<NA, 256, 0, stream>>>(adp, attn2, eqn_g, eqn_b, out_adp, NA);
}
